// MultiheadAttention_62173946576947
// MI455X (gfx1250) — matrix operand order repaired and checked by reading, not compiled, not run on hardware
//
#include <hip/hip_runtime.h>
#include <hip/hip_bf16.h>

// ---------------------------------------------------------------------------
// MHA for B=2, S=2048, D=1024, H=16, DK=64 on gfx1250 (CDNA5, wave32, WMMA).
// Pipeline: 3x GEMM (f32 in -> f16 QKV) -> flash attention (f16 WMMA, f32 acc,
// online softmax, sph multiplier + key-padding mask) -> GEMM (f32 out).
// All matmuls use v_wmma_f32_16x16x32_f16. All staging is vectorized
// (float4 / 8x f16 = b128 global loads), sph tile goes through LDS.
// ---------------------------------------------------------------------------

static constexpr int B_  = 2;
static constexpr int S_  = 2048;
static constexpr int D_  = 1024;
static constexpr int H_  = 16;
static constexpr int DK_ = 64;

typedef __attribute__((ext_vector_type(16))) _Float16 v16h;
typedef __attribute__((ext_vector_type(8)))  _Float16 v8h;
typedef __attribute__((ext_vector_type(4)))  _Float16 v4h;
typedef __attribute__((ext_vector_type(8)))  float    v8f;

union AFragU { v16h v; v8h p[2]; };

// WMMA 16x16x32 f16 -> f32 accum. 8 args per probe-confirmed signature.
__device__ inline v8f wmma_f16(v16h a, v16h b, v8f c) {
  return __builtin_amdgcn_wmma_f32_16x16x32_f16(false, a, false, b,
                                                (short)0, c, false, false);
}

// A-matrix fragment (16 rows x 32 K), row-major source with `stride` halves.
// ISA layout: lane L<16 holds row L, K = {0..7,16..23}; lanes 16-31 hold
// row L-16, K = {8..15,24..31}.  Two contiguous 8-half (16B) loads per lane.
__device__ inline v16h load_a_frag(const _Float16* tile, int stride, int lane) {
  int row = lane & 15;
  int h8  = (lane >> 4) * 8;
  AFragU f;
  f.p[0] = *(const v8h*)(tile + (size_t)row * stride + h8);
  f.p[1] = *(const v8h*)(tile + (size_t)row * stride + 16 + h8);
  return f.v;
}

// B-matrix fragment (32 K x 16 N) from [k][n] tile with `stride` halves.
// ISA layout: lane L holds column N = L&15; lanes 0-15 K = {0..7,16..23}, lanes 16-31 K = {8..15,24..31}.
__device__ inline v16h load_b_frag(const _Float16* tile, int stride, int lane) {
  int n  = lane & 15;
  int h8 = (lane >> 4) * 8;
  v16h b;
  for (int s = 0; s < 16; ++s) {
    int k = (s < 8) ? (h8 + s) : (16 + h8 + (s - 8));
    b[s] = tile[(size_t)k * stride + n];
  }
  return b;
}

// ---------------------------------------------------------------------------
// GEMM: out[M,N] = X[M,K] @ W[N,K]^T + bias[N].
// Block tile 64x128, K-step 32, 128 threads = 4 waves, each wave 32x64
// (2x4 WMMA tiles). A/W staged via float4 (b128) loads.
// ---------------------------------------------------------------------------
template <bool OUT_F16>
__global__ __launch_bounds__(128) void gemm_bias_kernel(
    const float* __restrict__ X, const float* __restrict__ W,
    const float* __restrict__ bias, _Float16* __restrict__ outh,
    float* __restrict__ outf, int M, int N, int K) {
  constexpr int BM = 64, BN = 128, BK = 32;
  __shared__ __align__(32) _Float16 Ah[BM * BK];  // [row][k]   stride 32
  __shared__ __align__(32) _Float16 Wt[BK * BN];  // [k][n]     stride 128

  const int t = threadIdx.x;
  const int lane = t & 31, wave = t >> 5;
  const int wm = (wave >> 1) * 32, wn = (wave & 1) * 64;
  const int m0 = blockIdx.y * BM, n0 = blockIdx.x * BN;

  v8f acc[2][4] = {};

  for (int k0 = 0; k0 < K; k0 += BK) {
    __syncthreads();
    // Stage A tile: 64x32 f32 = 512 float4, 4 per thread, b128 loads.
    for (int i = 0; i < 4; ++i) {
      int idx = i * 128 + t;
      int r = idx >> 3, c4 = (idx & 7) << 2;  // 8 float4 per row
      float4 x = *(const float4*)(X + (size_t)(m0 + r) * K + (k0 + c4));
      v4h hx = { (_Float16)x.x, (_Float16)x.y, (_Float16)x.z, (_Float16)x.w };
      *(v4h*)(Ah + r * BK + c4) = hx;
    }
    // Stage W tile transposed: 128 n-rows x 32 k = 1024 float4, 8 per thread.
    for (int i = 0; i < 8; ++i) {
      int idx = i * 128 + t;
      int r = idx >> 3, c4 = (idx & 7) << 2;  // r = n row, c4 = k
      float4 w = *(const float4*)(W + (size_t)(n0 + r) * K + (k0 + c4));
      Wt[(c4 + 0) * BN + r] = (_Float16)w.x;
      Wt[(c4 + 1) * BN + r] = (_Float16)w.y;
      Wt[(c4 + 2) * BN + r] = (_Float16)w.z;
      Wt[(c4 + 3) * BN + r] = (_Float16)w.w;
    }
    __syncthreads();

    v16h a0 = load_a_frag(Ah + (wm + 0) * BK, BK, lane);
    v16h a1 = load_a_frag(Ah + (wm + 16) * BK, BK, lane);
    for (int ni = 0; ni < 4; ++ni) {
      v16h bfr = load_b_frag(Wt + wn + ni * 16, BN, lane);
      acc[0][ni] = wmma_f16(a0, bfr, acc[0][ni]);
      acc[1][ni] = wmma_f16(a1, bfr, acc[1][ni]);
    }
  }

  // Epilogue: C layout — VGPR r: lanes 0-15 -> M=r, lanes 16-31 -> M=r+8; N=lane&15.
  const int half = lane >> 4, nl = lane & 15;
  for (int mi = 0; mi < 2; ++mi)
    for (int ni = 0; ni < 4; ++ni)
      for (int r = 0; r < 8; ++r) {
        int m = m0 + wm + mi * 16 + r + 8 * half;
        int n = n0 + wn + ni * 16 + nl;
        float v = acc[mi][ni][r] + bias[n];
        if (OUT_F16)
          outh[(size_t)m * N + n] = (_Float16)v;
        else
          outf[(size_t)m * N + n] = v;
      }
}

// ---------------------------------------------------------------------------
// Flash attention. Grid: (S/64, B*H); 128 threads = 4 waves, wave owns 16 q rows.
// scores = (Q K^T / 8) * sph, masked keys -> -1e30, online softmax, O = P V.
// K/V/sph tiles staged into LDS with b128 loads each 64-key step.
// ---------------------------------------------------------------------------
__global__ __launch_bounds__(128) void attention_kernel(
    const _Float16* __restrict__ qh, const _Float16* __restrict__ kh,
    const _Float16* __restrict__ vh, const float* __restrict__ sph,
    const unsigned char* __restrict__ mask, float* __restrict__ attn) {
  constexpr int KSTR = 80;  // LDS row stride (halves): 32B-aligned, conflict-skewed
  __shared__ __align__(32) _Float16 KT[64 * KSTR];   // [d][kk] (transposed)
  __shared__ __align__(32) _Float16 Vl[64 * KSTR];   // [kk][d]
  __shared__ __align__(32) _Float16 Pl[4][16 * 64];  // per-wave P staging
  __shared__ __align__(16) float    SPH[64 * 64];    // sph tile [q][kk]

  const int t = threadIdx.x, lane = t & 31, wave = t >> 5;
  const int half = lane >> 4, nl = lane & 15;
  const int bh = blockIdx.y;
  const int b = bh / H_, h = bh % H_;
  const int q0 = blockIdx.x * 64;
  const int qw = q0 + wave * 16;  // this wave's first q row
  const size_t hoff = (size_t)h * DK_;

  // Q fragments for dk 0..31 and 32..63, loaded straight from global.
  const _Float16* qbase = qh + ((size_t)b * S_ + qw) * D_ + hoff;
  v16h qa0 = load_a_frag(qbase, D_, lane);
  v16h qa1 = load_a_frag(qbase + 32, D_, lane);

  float mrow[8], lrow[8];
  for (int r = 0; r < 8; ++r) { mrow[r] = -1e30f; lrow[r] = 0.0f; }
  v8f O[4] = {};

  for (int kk0 = 0; kk0 < S_; kk0 += 64) {
    __syncthreads();
    // Stage K (transposed) + V: 64x64 f16 = 512 v8h each, 4 per thread.
    for (int i = 0; i < 4; ++i) {
      int idx = i * 128 + t;
      int kk = idx >> 3, d8 = (idx & 7) << 3;  // 8 v8h per key row
      size_t g = ((size_t)b * S_ + kk0 + kk) * D_ + hoff + d8;
      v8h kv = *(const v8h*)(kh + g);
      v8h vv = *(const v8h*)(vh + g);
      *(v8h*)(Vl + kk * KSTR + d8) = vv;
      for (int j = 0; j < 8; ++j) KT[(d8 + j) * KSTR + kk] = kv[j];
    }
    // Stage sph tile: 64x64 f32 = 1024 float4, 8 per thread, b128 loads.
    for (int i = 0; i < 8; ++i) {
      int idx = i * 128 + t;
      int q = idx >> 4, kk4 = (idx & 15) << 2;  // 16 float4 per q row
      *(float4*)(SPH + q * 64 + kk4) =
          *(const float4*)(sph + ((size_t)b * S_ + q0 + q) * S_ + kk0 + kk4);
    }
    __syncthreads();

    // Scores: 4 n-tiles of 16 keys, each = 2 WMMAs over dk.
    v8f sc[4];
    for (int nt = 0; nt < 4; ++nt) {
      v8f c = {};
      c = wmma_f16(qa0, load_b_frag(KT + nt * 16, KSTR, lane), c);
      c = wmma_f16(qa1, load_b_frag(KT + 32 * KSTR + nt * 16, KSTR, lane), c);
      sc[nt] = c;
    }

    // Elementwise: scale 1/sqrt(64), * sph (from LDS), key-padding mask.
    const int qlw = wave * 16 + 8 * half;  // local q row base for this lane half
    for (int nt = 0; nt < 4; ++nt) {
      int kkl = nt * 16 + nl;
      unsigned char mk = mask[(size_t)b * S_ + kk0 + kkl];
      for (int r = 0; r < 8; ++r) {
        float spv = SPH[(qlw + r) * 64 + kkl];
        float s = sc[nt][r] * 0.125f * spv;
        sc[nt][r] = mk ? -1e30f : s;
      }
    }

    // Online softmax per row (rows r / r+8 live in lane halves; reduce over 16 lanes).
    for (int r = 0; r < 8; ++r) {
      float mx = fmaxf(fmaxf(sc[0][r], sc[1][r]), fmaxf(sc[2][r], sc[3][r]));
      mx = fmaxf(mx, __shfl_xor(mx, 1, 32));
      mx = fmaxf(mx, __shfl_xor(mx, 2, 32));
      mx = fmaxf(mx, __shfl_xor(mx, 4, 32));
      mx = fmaxf(mx, __shfl_xor(mx, 8, 32));
      float mnew = fmaxf(mrow[r], mx);
      float corr = __expf(mrow[r] - mnew);
      mrow[r] = mnew;
      float rs = 0.0f;
      for (int nt = 0; nt < 4; ++nt) {
        float p = __expf(sc[nt][r] - mnew);
        sc[nt][r] = p;
        rs += p;
      }
      rs += __shfl_xor(rs, 1, 32);
      rs += __shfl_xor(rs, 2, 32);
      rs += __shfl_xor(rs, 4, 32);
      rs += __shfl_xor(rs, 8, 32);
      lrow[r] = lrow[r] * corr + rs;
      for (int dt = 0; dt < 4; ++dt) O[dt][r] *= corr;
    }

    // Restage P (16x64, f16) into per-wave LDS as an A-matrix source.
    _Float16* P = &Pl[wave][0];
    for (int nt = 0; nt < 4; ++nt)
      for (int r = 0; r < 8; ++r)
        P[(r + 8 * half) * 64 + nt * 16 + nl] = (_Float16)sc[nt][r];

    v16h pa0 = load_a_frag(P, 64, lane);       // keys 0..31 of this tile
    v16h pa1 = load_a_frag(P + 32, 64, lane);  // keys 32..63

    // O += P V  (two kk blocks x four 16-wide d tiles).
    for (int dt = 0; dt < 4; ++dt) {
      O[dt] = wmma_f16(pa0, load_b_frag(Vl + dt * 16, KSTR, lane), O[dt]);
      O[dt] = wmma_f16(pa1, load_b_frag(Vl + 32 * KSTR + dt * 16, KSTR, lane), O[dt]);
    }
  }

  // Write O / l as f32 into attn[b, q, h*DK + d].
  for (int dt = 0; dt < 4; ++dt)
    for (int r = 0; r < 8; ++r) {
      int qg = qw + r + 8 * half;
      int d = dt * 16 + nl;
      attn[((size_t)b * S_ + qg) * D_ + hoff + d] = O[dt][r] / lrow[r];
    }
}

// ---------------------------------------------------------------------------
// Launch: 3 projection GEMMs -> attention -> output GEMM. Workspace:
//   [0, 8MB)   Q f16   [8,16MB) K f16   [16,24MB) V f16   [24,40MB) attn f32
// ---------------------------------------------------------------------------
extern "C" void kernel_launch(void* const* d_in, const int* in_sizes, int n_in,
                              void* d_out, int out_size, void* d_ws,
                              size_t ws_size, hipStream_t stream) {
  const float* query = (const float*)d_in[0];
  const float* key   = (const float*)d_in[1];
  const float* value = (const float*)d_in[2];
  const float* sph   = (const float*)d_in[3];
  const unsigned char* mask = (const unsigned char*)d_in[4];  // jax bool = 1 byte
  const float* Wq = (const float*)d_in[5];
  const float* bq = (const float*)d_in[6];
  const float* Wk = (const float*)d_in[7];
  const float* bk = (const float*)d_in[8];
  const float* Wv = (const float*)d_in[9];
  const float* bv = (const float*)d_in[10];
  const float* Wo = (const float*)d_in[11];
  const float* bo = (const float*)d_in[12];

  const int M = B_ * S_, N = D_, K = D_;
  const size_t NE = (size_t)B_ * S_ * D_;

  _Float16* qh = (_Float16*)d_ws;
  _Float16* kh = qh + NE;
  _Float16* vh = kh + NE;
  float* attn  = (float*)(vh + NE);

  dim3 gblk(N / 128, M / 64);  // (8, 64)
  dim3 gthr(128);

  gemm_bias_kernel<true><<<gblk, gthr, 0, stream>>>(query, Wq, bq, qh, nullptr, M, N, K);
  gemm_bias_kernel<true><<<gblk, gthr, 0, stream>>>(key,   Wk, bk, kh, nullptr, M, N, K);
  gemm_bias_kernel<true><<<gblk, gthr, 0, stream>>>(value, Wv, bv, vh, nullptr, M, N, K);

  dim3 ablk(S_ / 64, B_ * H_);  // (32, 32)
  attention_kernel<<<ablk, gthr, 0, stream>>>(qh, kh, vh, sph, mask, attn);

  gemm_bias_kernel<false><<<gblk, gthr, 0, stream>>>(attn, Wo, bo, nullptr,
                                                     (float*)d_out, M, N, K);
}
